// GCN_60653528154601
// MI455X (gfx1250) — matrix operand order repaired and checked by reading, not compiled, not run on hardware
//
#include <hip/hip_runtime.h>
#include <hip/hip_bf16.h>

// ---------------------------------------------------------------------------
// GCN forward for MI455X (gfx1250, wave32).
//   Dense GEMMs      -> v_wmma_f32_16x16x32_f16 (fp32 accum), weights
//                       pre-packed once into B-fragment f16 layout
//   Edge aggregation -> wave-per-edge: scalar index loads + float4 gather
//                       + global fp32 atomic adds (L2-resident)
//   Pool             -> int-trick atomicMax (post-ReLU values are >= 0)
// ---------------------------------------------------------------------------

#define N_NODES   50000
#define N_EDGES   800000
#define F_IN      128
#define HID       128
#define N_CLASSES 10
#define N_GRAPHS  256

typedef __attribute__((ext_vector_type(16))) _Float16 v16h;
typedef __attribute__((ext_vector_type(8)))  float    v8f;

__device__ __forceinline__ void atomAddF(float* p, float v) {
  unsafeAtomicAdd(p, v);   // lowers to global_atomic_add_f32 on gfx1250
}

// ---------------- degree / norm ----------------
__global__ void k_deg_init(float* deg) {
  int i = blockIdx.x * blockDim.x + threadIdx.x;
  if (i < N_NODES) deg[i] = 1.0f;                 // self-loop
}

__global__ void k_deg_count(const int* __restrict__ dst, float* deg) {
  int e = blockIdx.x * blockDim.x + threadIdx.x;
  if (e < N_EDGES) atomAddF(&deg[dst[e]], 1.0f);
}

__global__ void k_rsqrt(float* deg) {
  int i = blockIdx.x * blockDim.x + threadIdx.x;
  if (i < N_NODES) deg[i] = rsqrtf(deg[i]);       // deg >= 1 always
}

// ---------------- weight pre-pack into B-fragment f16 layout ----------------
// For column tile c (0..7), K-chunk kc (0..3), lane (0..31), elem e (0..15):
//   B 32x16 f16 fragment: col = lane&15, K = (lane<16 ? 0 : 8) + (e<8 ? e : e+8)  (+kc*32)
// Packed order: [((c*4+kc)*32 + lane)*16 + e]  -> one v16h per (tile,chunk,lane)
__global__ void k_pack_w(const float* __restrict__ W, _Float16* __restrict__ out) {
  int tid = blockIdx.x * blockDim.x + threadIdx.x;   // 0..1023
  if (tid >= 8 * 4 * 32) return;
  const int c    = tid >> 7;
  const int kc   = (tid >> 5) & 3;
  const int lane = tid & 31;
  const int col  = c * 16 + (lane & 15);
  const int koff = kc * 32 + ((lane < 16) ? 0 : 8);
#pragma unroll
  for (int e = 0; e < 16; ++e)
    out[tid * 16 + e] = (_Float16)W[(koff + ((e < 8) ? e : e + 8)) * 128 + col];
}

// ---------------- WMMA GEMM: out[M x 128] = A[M x 128] @ W[128 x 128] -------
// One wave per 16x16 output tile; block = 256 threads = 8 waves = all 8
// column tiles of one 16-row stripe. M is a multiple of 16 (3125*16, 16*16).
// B comes from the packed f16 buffer (coalesced v16h loads). Optional row
// scale (D^-1/2 fold-in), bias, relu epilogues.
__global__ void k_wmma_gemm128(const float* __restrict__ A,
                               const _Float16* __restrict__ Wp,  // packed
                               const float* __restrict__ rowScale, // or null
                               const float* __restrict__ bias,     // or null
                               float* __restrict__ out,
                               int relu) {
  const int wave = threadIdx.x >> 5;      // 0..7 -> column tile
  const int lane = threadIdx.x & 31;
  const int row0 = blockIdx.x * 16;
  const int m    = lane & 15;             // A row within tile
  const int hiA  = (lane < 16) ? 0 : 8;   // A-frag K interleave (ISA 7.12.2)

  const v16h* Bp = (const v16h*)Wp + (wave * 4) * 32 + lane;

  v8f acc = {};
#pragma unroll
  for (int kc = 0; kc < 4; ++kc) {        // K = 128 in chunks of 32
    const int kbase = kc * 32;
    v16h a;
#pragma unroll
    for (int e = 0; e < 16; ++e) {
      const int v = e >> 1, j = e & 1;
      // A 16x32 f16 fragment: v<4 -> K = hi + 2v+j ; v>=4 -> 16+hi+2(v-4)+j
      const int ka = (v < 4) ? (hiA + 2 * v + j) : (16 + hiA + 2 * (v - 4) + j);
      a[e] = (_Float16)A[(row0 + m) * 128 + kbase + ka];
    }
    const v16h b = Bp[kc * 32];           // coalesced 32B packed-fragment load
    acc = __builtin_amdgcn_wmma_f32_16x16x32_f16(
        /*neg_a=*/false, a, /*neg_b=*/false, b,
        /*c_mod=*/(short)0, acc, /*reuse_a=*/false, /*reuse_b=*/false);
  }

  // C/D layout: lane<16 -> rows row0+0..7, lane>=16 -> rows row0+8..15
  const int col   = wave * 16 + (lane & 15);
  const int rbase = row0 + ((lane < 16) ? 0 : 8);
#pragma unroll
  for (int v = 0; v < 8; ++v) {
    float x = acc[v];
    if (rowScale) x *= rowScale[rbase + v];
    if (bias)     x += bias[col];
    if (relu)     x = fmaxf(x, 0.0f);
    out[(rbase + v) * 128 + col] = x;
  }
}

// ---------------- self-loop init: buf[i] = tmp[i] * dinv[i] -----------------
// (tmp already carries one dinv factor from the GEMM epilogue)
__global__ void k_selfinit(const float* __restrict__ tmp,
                           const float* __restrict__ dinv,
                           float* __restrict__ buf) {
  int idx = blockIdx.x * blockDim.x + threadIdx.x;
  if (idx < N_NODES * HID) buf[idx] = tmp[idx] * dinv[idx >> 7];
}

// ---------------- edge scatter: buf[d] += tmp[s] * dinv[d] ------------------
// One wave per edge: indices are wave-uniform -> scalar loads; each lane
// moves one float4 (32 lanes * 4 = full 128-float row).
__global__ void k_edge_scatter(const float* __restrict__ tmp,
                               const int* __restrict__ src,
                               const int* __restrict__ dst,
                               const float* __restrict__ dinv,
                               float* __restrict__ buf) {
  const int e = blockIdx.x * 8 + (threadIdx.x >> 5);  // 8 waves per block
  if (e >= N_EDGES) return;
  const int q = threadIdx.x & 31;
  const int s = src[e], d = dst[e];                   // wave-uniform -> SGPR
  const float w = dinv[d];
  const float4 v = ((const float4*)(tmp + (long long)s * 128))[q];
  float* o = buf + (long long)d * 128 + q * 4;
  atomAddF(o + 0, v.x * w);
  atomAddF(o + 1, v.y * w);
  atomAddF(o + 2, v.z * w);
  atomAddF(o + 3, v.w * w);
}

// ---------------- bias + relu (post-aggregation) ----------------------------
__global__ void k_bias_relu(float* __restrict__ buf, const float* __restrict__ b) {
  int idx = blockIdx.x * blockDim.x + threadIdx.x;
  if (idx < N_NODES * HID) buf[idx] = fmaxf(buf[idx] + b[idx & 127], 0.0f);
}

__global__ void k_zero(float* p, int n) {
  int i = blockIdx.x * blockDim.x + threadIdx.x;
  if (i < n) p[i] = 0.0f;
}

// ---------------- global max pool (values >= 0 post-ReLU) -------------------
__global__ void k_pool(const float* __restrict__ buf,
                       const int* __restrict__ batch,
                       float* __restrict__ g) {
  int idx = blockIdx.x * blockDim.x + threadIdx.x;
  if (idx >= N_NODES * HID) return;
  const int i = idx >> 7, f = idx & 127;
  atomicMax((int*)&g[batch[i] * HID + f], __float_as_int(buf[idx]));
}

// ---------------- final 128 -> 10 linear ------------------------------------
__global__ void k_linear(const float* __restrict__ g2,
                         const float* __restrict__ w,
                         const float* __restrict__ b,
                         float* __restrict__ out) {
  int tid = blockIdx.x * blockDim.x + threadIdx.x;
  if (tid >= N_GRAPHS * N_CLASSES) return;
  const int gi = tid / N_CLASSES, c = tid % N_CLASSES;
  float acc = b[c];
  const float* row = g2 + gi * HID;
#pragma unroll 4
  for (int k = 0; k < HID; ++k) acc += row[k] * w[k * N_CLASSES + c];
  out[tid] = acc;
}

// ---------------------------------------------------------------------------
extern "C" void kernel_launch(void* const* d_in, const int* in_sizes, int n_in,
                              void* d_out, int out_size, void* d_ws, size_t ws_size,
                              hipStream_t stream) {
  const float* x     = (const float*)d_in[0];
  const int*   edge  = (const int*)d_in[1];
  const int*   src   = edge;
  const int*   dst   = edge + N_EDGES;
  const int*   batch = (const int*)d_in[2];
  const float* W1 = (const float*)d_in[3];  const float* b1 = (const float*)d_in[4];
  const float* W2 = (const float*)d_in[5];  const float* b2 = (const float*)d_in[6];
  const float* W3 = (const float*)d_in[7];  const float* b3 = (const float*)d_in[8];
  const float* fc1w = (const float*)d_in[9];  const float* fc1b = (const float*)d_in[10];
  const float* linw = (const float*)d_in[11]; const float* linb = (const float*)d_in[12];
  float* out = (float*)d_out;

  // workspace carve-up (all regions fully overwritten every call)
  float* tmp  = (float*)d_ws;                    // [N_NODES*HID]
  float* buf  = tmp  + (size_t)N_NODES * HID;    // [N_NODES*HID]
  float* dinv = buf  + (size_t)N_NODES * HID;    // [N_NODES]
  float* g    = dinv + N_NODES;                  // [N_GRAPHS*HID]
  float* g2   = g    + (size_t)N_GRAPHS * HID;   // [N_GRAPHS*HID]
  _Float16* wp = (_Float16*)(g2 + (size_t)N_GRAPHS * HID); // 4 x [128*128] f16

  const int B = 256;
  const int nNH = N_NODES * HID;
  const int gN  = (N_NODES + B - 1) / B;
  const int gE  = (N_EDGES + B - 1) / B;
  const int gNH = (nNH + B - 1) / B;

  // degree -> dinv
  k_deg_init<<<gN, B, 0, stream>>>(dinv);
  k_deg_count<<<gE, B, 0, stream>>>(dst, dinv);
  k_rsqrt<<<gN, B, 0, stream>>>(dinv);

  // pack all weight matrices into B-fragment f16 layout (once per launch)
  _Float16* wp1 = wp;             _Float16* wp2 = wp + 16384;
  _Float16* wp3 = wp + 2 * 16384; _Float16* wpf = wp + 3 * 16384;
  k_pack_w<<<4, B, 0, stream>>>(W1, wp1);
  k_pack_w<<<4, B, 0, stream>>>(W2, wp2);
  k_pack_w<<<4, B, 0, stream>>>(W3, wp3);
  k_pack_w<<<4, B, 0, stream>>>(fc1w, wpf);

  const _Float16* Wps[3] = {wp1, wp2, wp3};
  const float*    bs[3]  = {b1, b2, b3};
  const float* in = x;
  for (int layer = 0; layer < 3; ++layer) {
    // tmp = (in @ W) * dinv[row]   (one dinv factor folded into the epilogue)
    k_wmma_gemm128<<<N_NODES / 16, 256, 0, stream>>>(in, Wps[layer], dinv,
                                                     nullptr, tmp, 0);
    // buf = self-loop term, then scatter-add edge messages, then bias+relu
    k_selfinit<<<gNH, B, 0, stream>>>(tmp, dinv, buf);
    k_edge_scatter<<<(N_EDGES + 7) / 8, B, 0, stream>>>(tmp, src, dst, dinv, buf);
    k_bias_relu<<<gNH, B, 0, stream>>>(buf, bs[layer]);
    in = buf;
  }

  // global max pool over sorted batch ids
  k_zero<<<(N_GRAPHS * HID + B - 1) / B, B, 0, stream>>>(g, N_GRAPHS * HID);
  k_pool<<<gNH, B, 0, stream>>>(buf, batch, g);

  // g2 = relu(g @ fc1_w + fc1_b)   (256 rows -> 16 row tiles)
  k_wmma_gemm128<<<N_GRAPHS / 16, 256, 0, stream>>>(g, wpf, nullptr, fc1b, g2, 1);

  // out = g2 @ lin_w + lin_b
  k_linear<<<(N_GRAPHS * N_CLASSES + B - 1) / B, B, 0, stream>>>(g2, linw, linb, out);
}
